// DgcnNetwork_83726092468415
// MI455X (gfx1250) — matrix operand order repaired and checked by reading, not compiled, not run on hardware
//
#include <hip/hip_runtime.h>

// ---------------------------------------------------------------------------
// DGCNN EdgeConv network for MI455X (gfx1250, wave32, WMMA, async-to-LDS).
// Bandwidth-bound (~10 GB activation traffic vs 23.3 TB/s HBM):
//  - activations/weights bf16; GEMMs use v_wmma_f32_16x16x32_bf16
//  - A tiles staged to LDS with global_load_async_to_lds_b128 (ASYNCcnt,
//    double-buffered, s_wait_asynccnt) -- no VGPR round trip
//  - LDS row pitch 80B -> conflict-free ds_load_b128 fragment reads
//  - 4 independent A fragments preloaded -> 4 back-to-back WMMAs per k-step
//  - bias+ReLU+BN statistics fused into GEMM epilogue (shfl + f32 atomics)
//  - BN apply fused with bf16 down-convert; conv layer-2 apply fused with
//    the segment_sum scatter (global_atomic_add_f32)
// Assumes M (=E or N) multiple of 16 and Nout multiple of 128 for the WMMA
// path -- true here (E=800000, N=50000, Nout in {128,256,1024}).
// ---------------------------------------------------------------------------

typedef __attribute__((ext_vector_type(16))) __bf16 v16bf;
typedef __attribute__((ext_vector_type(8)))  __bf16 v8bf;
typedef __attribute__((ext_vector_type(4)))  __bf16 v4bf;
typedef __attribute__((ext_vector_type(8)))  float  v8f;

#define EPS_BN 1e-5f
#define MTILE     64   // rows per block (4 subtiles of 16)
#define LDS_PITCH 40   // bf16 elements per LDS row (80B): r*20 mod 64 distinct
                       // for r=0..15 -> conflict-free b128 reads

// ---------------------------------------------------------------------------
// Utility kernels
// ---------------------------------------------------------------------------
__global__ void zero_f32(float* __restrict__ p, long n) {
    long t = blockIdx.x * (long)blockDim.x + threadIdx.x;
    if (t < n) p[t] = 0.0f;
}

// fp32 -> bf16 weight convert with optional zero row padding (conv1: 6 -> 32)
__global__ void cvt_pad_bf16(const float* __restrict__ src, __bf16* __restrict__ dst,
                             int rows, int rowsPad, int cols) {
    long t = blockIdx.x * (long)blockDim.x + threadIdx.x;
    long total = (long)rowsPad * cols;
    if (t >= total) return;
    int r = (int)(t / cols);
    int c = (int)(t % cols);
    dst[t] = (r < rows) ? (__bf16)src[(long)r * cols + c] : (__bf16)0.0f;
}

// ---------------------------------------------------------------------------
// Edge feature gather: e = [x_i, x_j - x_i]  (i = dst, j = src)
// ---------------------------------------------------------------------------
// conv1: F=3 inputs, zero-padded to K=32 so the WMMA GEMM handles it.
__global__ void gather_conv1(const float* __restrict__ x,
                             const int* __restrict__ srcI, const int* __restrict__ dstI,
                             __bf16* __restrict__ outp, int E) {
    int e = blockIdx.x * blockDim.x + threadIdx.x;
    if (e >= E) return;
    int s = srcI[e], d = dstI[e];
    float xi0 = x[d * 3 + 0], xi1 = x[d * 3 + 1], xi2 = x[d * 3 + 2];
    float xj0 = x[s * 3 + 0], xj1 = x[s * 3 + 1], xj2 = x[s * 3 + 2];
    union { __bf16 b[32]; v8bf v[4]; } u;
#pragma unroll
    for (int k = 0; k < 32; k++) u.b[k] = (__bf16)0.0f;
    u.b[0] = (__bf16)xi0; u.b[1] = (__bf16)xi1; u.b[2] = (__bf16)xi2;
    u.b[3] = (__bf16)(xj0 - xi0); u.b[4] = (__bf16)(xj1 - xi1); u.b[5] = (__bf16)(xj2 - xi2);
    v8bf* o = (v8bf*)(outp + (size_t)e * 32);
    o[0] = u.v[0]; o[1] = u.v[1]; o[2] = u.v[2]; o[3] = u.v[3];
}

// conv2/3: C=128 node features -> [E, 256] bf16, 4 channels per thread.
__global__ void gather_edges128(const float* __restrict__ x,
                                const int* __restrict__ srcI, const int* __restrict__ dstI,
                                __bf16* __restrict__ outp, int E) {
    long t = blockIdx.x * (long)blockDim.x + threadIdx.x;
    if (t >= (long)E * 32) return;
    int e = (int)(t >> 5);
    int q = ((int)t & 31) * 4;
    int s = srcI[e], d = dstI[e];
    float4 a = *(const float4*)(x + (size_t)d * 128 + q);
    float4 b = *(const float4*)(x + (size_t)s * 128 + q);
    v4bf o1 = {(__bf16)a.x, (__bf16)a.y, (__bf16)a.z, (__bf16)a.w};
    v4bf o2 = {(__bf16)(b.x - a.x), (__bf16)(b.y - a.y),
               (__bf16)(b.z - a.z), (__bf16)(b.w - a.w)};
    *(v4bf*)(outp + (size_t)e * 256 + q)       = o1;
    *(v4bf*)(outp + (size_t)e * 256 + 128 + q) = o2;
}

// concat [x1|x2|x3] (N x 384) and convert to bf16
__global__ void concat_cvt(const float* __restrict__ x1, const float* __restrict__ x2,
                           const float* __restrict__ x3, __bf16* __restrict__ outp, int Nn) {
    long t = blockIdx.x * (long)blockDim.x + threadIdx.x;
    if (t >= (long)Nn * 96) return;
    int node = (int)(t / 96);
    int c    = ((int)(t % 96)) * 4;
    const float* sp; int cc;
    if (c < 128)      { sp = x1; cc = c; }
    else if (c < 256) { sp = x2; cc = c - 128; }
    else              { sp = x3; cc = c - 256; }
    float4 v = *(const float4*)(sp + (size_t)node * 128 + cc);
    v4bf o = {(__bf16)v.x, (__bf16)v.y, (__bf16)v.z, (__bf16)v.w};
    *(v4bf*)(outp + (size_t)node * 384 + c) = o;
}

// ---------------------------------------------------------------------------
// WMMA GEMM: out[M,Nout] = relu(A[M,K]bf16 @ B[K,Nout]bf16 + bias), with
// fused per-column sum / sum-of-squares accumulation for BatchNorm stats.
// Block = 256 threads = 8 waves; block tile 64 rows x 128 cols.
// Each wave: 4 M-subtiles x 1 N-tile (4 accumulators, B fragment reused 4x).
// A staged global->LDS by async DMA (double buffered); B loaded direct
// (tiny, L2-resident) via an incremented pointer (no per-step mul).
// ---------------------------------------------------------------------------
__global__ __launch_bounds__(256)
void wmma_gemm_bias_relu_stats(const __bf16* __restrict__ A,
                               const __bf16* __restrict__ B,
                               const float*  __restrict__ bias,
                               float* __restrict__ out,
                               float* __restrict__ colsum,
                               float* __restrict__ colsumsq,
                               int M, int K, int Nout) {
    __shared__ __bf16 shA[2][MTILE * LDS_PITCH];   // 2 x 5120B

    const int lane = threadIdx.x & 31;
    const int wave = threadIdx.x >> 5;
    const int m0 = blockIdx.x * MTILE;
    const int n0 = blockIdx.y * 128 + wave * 16;

    // --- A staging: thread t copies 16B: row = t/4 (0..63), koff = (t%4)*8
    const int srow  = threadIdx.x >> 2;
    const int skoff = (threadIdx.x & 3) * 8;
    int growi = m0 + srow;
    if (growi > M - 1) growi = M - 1;              // clamp tail rows (M%64 != 0)
    const __bf16* gRow = A + (size_t)growi * K + skoff;
    const unsigned ldsDst[2] = {
        (unsigned)(size_t)&shA[0][srow * LDS_PITCH + skoff],
        (unsigned)(size_t)&shA[1][srow * LDS_PITCH + skoff]
    };

    const int hi   = lane >> 4;    // 0: K octets {0,16}; 1: {8,24}
    const int frow = lane & 15;

    v8f acc0 = {}, acc1 = {}, acc2 = {}, acc3 = {};

    // per-lane B pointer: column n0+frow, rows kk+hi*8 (+0..7, +16..23); += 32 rows / step
    const __bf16* Bp = B + (size_t)(hi * 8) * Nout + n0 + frow;
    const size_t bStep = (size_t)32 * Nout;

    union Af { v16bf v; v8bf h[2]; };
    auto kstep = [&](int cur) {
        v16bf b;
#pragma unroll
        for (int s = 0; s < 16; s++) b[s] = Bp[(size_t)(s < 8 ? s : s + 8) * Nout];
        Bp += bStep;
        const __bf16* base = &shA[cur][frow * LDS_PITCH + hi * 8];
        Af a0, a1, a2, a3;   // 4 independent fragments -> back-to-back WMMAs
        a0.h[0] = *(const v8bf*)(base + 0 * 16 * LDS_PITCH);
        a0.h[1] = *(const v8bf*)(base + 0 * 16 * LDS_PITCH + 16);
        a1.h[0] = *(const v8bf*)(base + 1 * 16 * LDS_PITCH);
        a1.h[1] = *(const v8bf*)(base + 1 * 16 * LDS_PITCH + 16);
        a2.h[0] = *(const v8bf*)(base + 2 * 16 * LDS_PITCH);
        a2.h[1] = *(const v8bf*)(base + 2 * 16 * LDS_PITCH + 16);
        a3.h[0] = *(const v8bf*)(base + 3 * 16 * LDS_PITCH);
        a3.h[1] = *(const v8bf*)(base + 3 * 16 * LDS_PITCH + 16);
        acc0 = __builtin_amdgcn_wmma_f32_16x16x32_bf16(false, a0.v, false, b, (short)0, acc0, false, false);
        acc1 = __builtin_amdgcn_wmma_f32_16x16x32_bf16(false, a1.v, false, b, (short)0, acc1, false, false);
        acc2 = __builtin_amdgcn_wmma_f32_16x16x32_bf16(false, a2.v, false, b, (short)0, acc2, false, false);
        acc3 = __builtin_amdgcn_wmma_f32_16x16x32_bf16(false, a3.v, false, b, (short)0, acc3, false, false);
    };

    const int steps = K / 32;
    // prologue DMA for buffer 0
    asm volatile("global_load_async_to_lds_b128 %0, %1, off"
                 :: "v"(ldsDst[0]), "v"(gRow) : "memory");
    // steady state: unconditional issue of next buffer, wait own DMA <= 1
    for (int i = 0; i < steps - 1; ++i) {
        const int cur = i & 1;
        asm volatile("global_load_async_to_lds_b128 %0, %1, off"
                     :: "v"(ldsDst[cur ^ 1]), "v"(gRow + (size_t)(i + 1) * 32) : "memory");
        asm volatile("s_wait_asynccnt 0x1" ::: "memory");  // buf[cur] DMA done (in-order)
        __syncthreads();                                    // all waves' DMA visible
        kstep(cur);
        __syncthreads();   // everyone done reading buf[cur] before it is re-filled
    }
    // peeled last step
    asm volatile("s_wait_asynccnt 0x0" ::: "memory");
    __syncthreads();
    kstep((steps - 1) & 1);

    // Epilogue: bias + ReLU, store fp32, fold column stats (one atomic pair
    // per column per 64-row block). Subtile validity at 16-row granularity.
    const int n = n0 + frow;
    const float bv = bias[n];
    float s = 0.0f, s2 = 0.0f;
#define EPILOG(sub, accv)                                                        \
    if (m0 + (sub)*16 < M) {                                                     \
        const int rbase = m0 + (sub)*16 + hi * 8;                                \
        _Pragma("unroll")                                                        \
        for (int i = 0; i < 8; i++) {                                            \
            float v = accv[i] + bv;                                              \
            v = v > 0.0f ? v : 0.0f;                                             \
            out[(size_t)(rbase + i) * Nout + n] = v;                             \
            s += v; s2 += v * v;                                                 \
        }                                                                        \
    }
    EPILOG(0, acc0)
    EPILOG(1, acc1)
    EPILOG(2, acc2)
    EPILOG(3, acc3)
#undef EPILOG
    // lanes l and l^16 hold the same column -> combine, then one atomic per col
    s  += __shfl_xor(s, 16);
    s2 += __shfl_xor(s2, 16);
    if (lane < 16) {
        atomicAdd(&colsum[n], s);
        atomicAdd(&colsumsq[n], s2);
    }
}

// ---------------------------------------------------------------------------
// BatchNorm finalize + apply
// ---------------------------------------------------------------------------
__global__ void bn_finalize(const float* __restrict__ sum, const float* __restrict__ sumsq,
                            const float* __restrict__ gamma, const float* __restrict__ beta,
                            float* __restrict__ scale, float* __restrict__ shift,
                            float invM, int Nout) {
    int c = blockIdx.x * blockDim.x + threadIdx.x;
    if (c >= Nout) return;
    float mu  = sum[c] * invM;
    float var = sumsq[c] * invM - mu * mu;      // biased variance (matches jnp.var)
    var = var > 0.0f ? var : 0.0f;
    float inv = rsqrtf(var + EPS_BN);
    float sc  = gamma[c] * inv;
    scale[c] = sc;
    shift[c] = beta[c] - mu * sc;
}

// y = scale[c]*h + shift[c], bf16 output (feeds next WMMA GEMM)
__global__ void bn_apply_cvt(const float* __restrict__ h,
                             const float* __restrict__ scale, const float* __restrict__ shift,
                             __bf16* __restrict__ outp, long total, int Nout) {
    long t = (blockIdx.x * (long)blockDim.x + threadIdx.x) * 4;
    if (t >= total) return;
    int c = (int)(t % Nout);
    float4 v = *(const float4*)(h + t);
    v4bf o = {(__bf16)(scale[c] * v.x + shift[c]),
              (__bf16)(scale[c + 1] * v.y + shift[c + 1]),
              (__bf16)(scale[c + 2] * v.z + shift[c + 2]),
              (__bf16)(scale[c + 3] * v.w + shift[c + 3])};
    *(v4bf*)(outp + t) = o;
}

// BN apply fused with segment_sum scatter onto destination nodes (C=128)
__global__ void bn_apply_scatter(const float* __restrict__ h,
                                 const float* __restrict__ scale, const float* __restrict__ shift,
                                 const int* __restrict__ dstI, float* __restrict__ xout, int E) {
    long t = blockIdx.x * (long)blockDim.x + threadIdx.x;
    if (t >= (long)E * 128) return;
    int e = (int)(t >> 7);
    int c = (int)t & 127;
    float v = scale[c] * h[t] + shift[c];
    atomicAdd(xout + (size_t)dstI[e] * 128 + c, v);
}

// ---------------------------------------------------------------------------
// Head: 128 -> 24 linear (fp32 weights) + log_softmax, one node per thread.
// ---------------------------------------------------------------------------
__global__ void head_logsoftmax(const __bf16* __restrict__ hin,
                                const float* __restrict__ W, const float* __restrict__ b,
                                float* __restrict__ outp, int Nn) {
    int node = blockIdx.x * blockDim.x + threadIdx.x;
    if (node >= Nn) return;
    float logits[24];
#pragma unroll
    for (int j = 0; j < 24; j++) logits[j] = b[j];
    const __bf16* hp = hin + (size_t)node * 128;
    for (int k = 0; k < 128; k++) {
        float xv = (float)hp[k];
        const float* wr = W + k * 24;
#pragma unroll
        for (int j = 0; j < 24; j++) logits[j] += xv * wr[j];
    }
    float mx = logits[0];
#pragma unroll
    for (int j = 1; j < 24; j++) mx = fmaxf(mx, logits[j]);
    float se = 0.0f;
#pragma unroll
    for (int j = 0; j < 24; j++) se += expf(logits[j] - mx);
    float lse = mx + logf(se);
#pragma unroll
    for (int j = 0; j < 24; j++) outp[(size_t)node * 24 + j] = logits[j] - lse;
}

// ---------------------------------------------------------------------------
// Host orchestration
// ---------------------------------------------------------------------------
extern "C" void kernel_launch(void* const* d_in, const int* in_sizes, int n_in,
                              void* d_out, int out_size, void* d_ws, size_t ws_size,
                              hipStream_t stream) {
    (void)n_in; (void)out_size; (void)ws_size;

    const float* x  = (const float*)d_in[0];
    const int*   ei = (const int*)d_in[1];
    const int Nn = in_sizes[0] / 3;   // 50000
    const int E  = in_sizes[1] / 2;   // 800000
    const int* srcI = ei;
    const int* dstI = ei + E;

    // params in setup_inputs() insertion order: conv1,conv2,conv3,lin1,head_bn,head_out
    const float* c1W1 = (const float*)d_in[2];  const float* c1b1 = (const float*)d_in[3];
    const float* c1g1 = (const float*)d_in[4];  const float* c1e1 = (const float*)d_in[5];
    const float* c1W2 = (const float*)d_in[6];  const float* c1b2 = (const float*)d_in[7];
    const float* c1g2 = (const float*)d_in[8];  const float* c1e2 = (const float*)d_in[9];
    const float* c2W1 = (const float*)d_in[10]; const float* c2b1 = (const float*)d_in[11];
    const float* c2g1 = (const float*)d_in[12]; const float* c2e1 = (const float*)d_in[13];
    const float* c2W2 = (const float*)d_in[14]; const float* c2b2 = (const float*)d_in[15];
    const float* c2g2 = (const float*)d_in[16]; const float* c2e2 = (const float*)d_in[17];
    const float* c3W1 = (const float*)d_in[18]; const float* c3b1 = (const float*)d_in[19];
    const float* c3g1 = (const float*)d_in[20]; const float* c3e1 = (const float*)d_in[21];
    const float* c3W2 = (const float*)d_in[22]; const float* c3b2 = (const float*)d_in[23];
    const float* c3g2 = (const float*)d_in[24]; const float* c3e2 = (const float*)d_in[25];
    const float* l1W  = (const float*)d_in[26]; const float* l1b  = (const float*)d_in[27];
    const float* l1g  = (const float*)d_in[28]; const float* l1e  = (const float*)d_in[29];
    const float* hAW  = (const float*)d_in[30]; const float* hAb  = (const float*)d_in[31];
    const float* hAg  = (const float*)d_in[32]; const float* hAe  = (const float*)d_in[33];
    const float* hBW  = (const float*)d_in[34]; const float* hBb  = (const float*)d_in[35];
    const float* hBg  = (const float*)d_in[36]; const float* hBe  = (const float*)d_in[37];
    const float* hoW  = (const float*)d_in[38]; const float* hob  = (const float*)d_in[39];

    // workspace carve-up
    char* ws = (char*)d_ws;
    size_t cur = 0;
    auto take = [&](size_t bytes) -> void* {
        void* p = ws + cur;
        cur = (cur + bytes + 255) & ~(size_t)255;
        return p;
    };
    __bf16* eBuf  = (__bf16*)take((size_t)E * 256 * 2);   // edge gather (max K=256)
    float*  hBuf  = (float*) take((size_t)E * 128 * 4);   // fp32 GEMM out (>= N*1024*4)
    __bf16* hnBuf = (__bf16*)take((size_t)E * 128 * 2);   // bf16 normalized (>= N*1024*2)
    float*  x1    = (float*) take((size_t)Nn * 128 * 4);
    float*  x2    = (float*) take((size_t)Nn * 128 * 4);
    float*  x3    = (float*) take((size_t)Nn * 128 * 4);
    __bf16* xcat  = (__bf16*)take((size_t)Nn * 384 * 2);
    float*  statsB = (float*)take(4 * 1024 * 4);          // sum|sumsq|scale|shift
    float* sumB   = statsB;
    float* sumsqB = statsB + 1024;
    float* scaleB = statsB + 2048;
    float* shiftB = statsB + 3072;
    __bf16* w_c1a = (__bf16*)take(32   * 128  * 2);
    __bf16* w_c1b = (__bf16*)take(128  * 128  * 2);
    __bf16* w_c2a = (__bf16*)take(256  * 128  * 2);
    __bf16* w_c2b = (__bf16*)take(128  * 128  * 2);
    __bf16* w_c3a = (__bf16*)take(256  * 128  * 2);
    __bf16* w_c3b = (__bf16*)take(128  * 128  * 2);
    __bf16* w_l1  = (__bf16*)take(384  * 1024 * 2);
    __bf16* w_hA  = (__bf16*)take(1024 * 256  * 2);
    __bf16* w_hB  = (__bf16*)take(256  * 128  * 2);

    auto cvtW = [&](const float* src, __bf16* dst, int rows, int rowsPad, int cols) {
        long tot = (long)rowsPad * cols;
        cvt_pad_bf16<<<(unsigned)((tot + 255) / 256), 256, 0, stream>>>(src, dst, rows, rowsPad, cols);
    };
    cvtW(c1W1, w_c1a, 6,   32,   128);
    cvtW(c1W2, w_c1b, 128, 128,  128);
    cvtW(c2W1, w_c2a, 256, 256,  128);
    cvtW(c2W2, w_c2b, 128, 128,  128);
    cvtW(c3W1, w_c3a, 256, 256,  128);
    cvtW(c3W2, w_c3b, 128, 128,  128);
    cvtW(l1W,  w_l1,  384, 384,  1024);
    cvtW(hAW,  w_hA,  1024, 1024, 256);
    cvtW(hBW,  w_hB,  256, 256,  128);

    auto gemm = [&](const __bf16* A, const __bf16* Bw, const float* bias, float* outp,
                    int M, int K, int Nout) {
        zero_f32<<<8, 256, 0, stream>>>(statsB, 2048);
        dim3 grid((unsigned)((M + MTILE - 1) / MTILE), (unsigned)(Nout / 128));
        wmma_gemm_bias_relu_stats<<<grid, 256, 0, stream>>>(A, Bw, bias, outp,
                                                            sumB, sumsqB, M, K, Nout);
    };
    auto finalize = [&](const float* g, const float* be, int M, int Nout) {
        bn_finalize<<<(unsigned)((Nout + 63) / 64), 64, 0, stream>>>(
            sumB, sumsqB, g, be, scaleB, shiftB, 1.0f / (float)M, Nout);
    };
    auto applyCvt = [&](const float* h, __bf16* o, long total, int Nout) {
        bn_apply_cvt<<<(unsigned)((total / 4 + 255) / 256), 256, 0, stream>>>(
            h, scaleB, shiftB, o, total, Nout);
    };

    auto run_conv = [&](const float* xin, bool first,
                        const __bf16* W1bf, int K1, const float* b1, const float* g1, const float* e1,
                        const __bf16* W2bf, const float* b2, const float* g2, const float* e2,
                        float* xoutp) {
        if (first)
            gather_conv1<<<(unsigned)((E + 255) / 256), 256, 0, stream>>>(xin, srcI, dstI, eBuf, E);
        else
            gather_edges128<<<(unsigned)(((long)E * 32 + 255) / 256), 256, 0, stream>>>(xin, srcI, dstI, eBuf, E);
        // layer 1
        gemm(eBuf, W1bf, b1, hBuf, E, K1, 128);
        finalize(g1, e1, E, 128);
        applyCvt(hBuf, hnBuf, (long)E * 128, 128);
        // layer 2
        gemm(hnBuf, W2bf, b2, hBuf, E, 128, 128);
        finalize(g2, e2, E, 128);
        // segment_sum: zero accumulator then fused BN-apply + scatter
        zero_f32<<<(unsigned)(((long)Nn * 128 + 255) / 256), 256, 0, stream>>>(xoutp, (long)Nn * 128);
        bn_apply_scatter<<<(unsigned)(((long)E * 128 + 255) / 256), 256, 0, stream>>>(
            hBuf, scaleB, shiftB, dstI, xoutp, E);
    };

    run_conv(x,  true,  w_c1a, 32,  c1b1, c1g1, c1e1, w_c1b, c1b2, c1g2, c1e2, x1);
    run_conv(x1, false, w_c2a, 256, c2b1, c2g1, c2e1, w_c2b, c2b2, c2g2, c2e2, x2);
    run_conv(x2, false, w_c3a, 256, c3b1, c3g1, c3e1, w_c3b, c3b2, c3g2, c3e2, x3);

    // lin1: [N,384] -> 1024
    concat_cvt<<<(unsigned)(((long)Nn * 96 + 255) / 256), 256, 0, stream>>>(x1, x2, x3, xcat, Nn);
    gemm(xcat, w_l1, l1b, hBuf, Nn, 384, 1024);
    finalize(l1g, l1e, Nn, 1024);
    applyCvt(hBuf, hnBuf, (long)Nn * 1024, 1024);

    // head_bn layer 0: 1024 -> 256
    gemm(hnBuf, w_hA, hAb, hBuf, Nn, 1024, 256);
    finalize(hAg, hAe, Nn, 256);
    applyCvt(hBuf, hnBuf, (long)Nn * 256, 256);

    // head_bn layer 1: 256 -> 128
    gemm(hnBuf, w_hB, hBb, hBuf, Nn, 256, 128);
    finalize(hBg, hBe, Nn, 128);
    applyCvt(hBuf, hnBuf, (long)Nn * 128, 128);

    // head_out 128 -> 24 + log_softmax
    head_logsoftmax<<<(unsigned)((Nn + 255) / 256), 256, 0, stream>>>(
        hnBuf, hoW, hob, (float*)d_out, Nn);
}
